// QuantumHybridResNet_31817117729345
// MI455X (gfx1250) — matrix operand order repaired and checked by reading, not compiled, not run on hardware
//
#include <hip/hip_runtime.h>

typedef __attribute__((ext_vector_type(16))) _Float16 v16h;
typedef __attribute__((ext_vector_type(8)))  _Float16 v8h;
typedef __attribute__((ext_vector_type(8)))  float    v8f;

#define BM 128
#define BN 64
#define KB 64

// ---------------------------------------------------------------------------
// Input: NCHW f32 -> NHWC f16  (hardcoded 64x3x224x224)
// ---------------------------------------------------------------------------
__global__ void cvt_nchw_to_nhwc_kernel(const float* __restrict__ src,
                                        _Float16* __restrict__ dst) {
  const int C = 3, HW = 224;
  int i = blockIdx.x * blockDim.x + threadIdx.x;
  if (i >= 64 * 3 * 224 * 224) return;
  int c = i % C;
  int t = i / C;
  int w = t % HW; t /= HW;
  int h = t % HW;
  int n = t / HW;
  dst[i] = (_Float16)src[(((long)n * C + c) * HW + h) * HW + w];
}

// W[K][C][R][S] f32 -> Wp[(r*S+s)*C + c][K] f16, zero pad rows crs >= CRS
__global__ void repack_weight_kernel(const float* __restrict__ W, _Float16* __restrict__ Wp,
                                     int K, int C, int R, int S, int CRSp) {
  int idx = blockIdx.x * blockDim.x + threadIdx.x;
  if (idx >= CRSp * K) return;
  int k = idx % K;
  int crs = idx / K;
  int CRS = C * R * S;
  float v = 0.0f;
  if (crs < CRS) {
    int c = crs % C;
    int rsi = crs / C;
    int s = rsi % S;
    int r = rsi / S;
    v = W[(((size_t)k * C + c) * R + r) * S + s];
  }
  Wp[(size_t)crs * K + k] = (_Float16)v;
}

// ---------------------------------------------------------------------------
// Implicit-GEMM conv (NHWC f16) + fused BN (+residual) (+ReLU), f32 accum.
//   M = Nb*P*Q (pixels == NHWC flattened pixel index), N = K, red = R*S*C
// FAST: C is a power of two >= 32 -> contiguous 32B channel runs, async A+B.
// KB=64: 8 WMMAs per barrier/asynccnt-wait.
// ---------------------------------------------------------------------------
struct ConvArgs {
  const _Float16* X;      // NHWC
  const _Float16* Wp;     // packed [CRSp][K]
  const float *gamma, *beta, *mean, *var;
  const _Float16* Res;    // NHWC residual or nullptr
  _Float16* Y;            // NHWC
  int Nb, C, H, W, K, P, Q, R, S, stride, pad, CRS, CRSp, relu;
  int clog2;              // log2(C) (FAST only)
  unsigned mulC, mulS;    // ceil(2^16 / C), ceil(2^16 / S)
};

template <bool FAST>
__global__ __launch_bounds__(256) void conv_wmma_kernel(ConvArgs a) {
  __shared__ __align__(32) _Float16 As[2 * BM * KB];   // 32 KB
  __shared__ __align__(32) _Float16 Bs[2 * KB * BN];   // 16 KB
  __shared__ float s_scale[BN];
  __shared__ float s_bias[BN];

  const int tid  = threadIdx.x;
  const int n0   = blockIdx.y * BN;
  const int m0   = blockIdx.x * BM;
  const int Mtot = a.Nb * a.P * a.Q;

  if (tid < BN) {
    int ko = n0 + tid;
    float sc = a.gamma[ko] * rsqrtf(a.var[ko] + 1e-5f);
    s_scale[tid] = sc;
    s_bias[tid]  = a.beta[ko] - a.mean[ko] * sc;
  }

  // ---- per-thread A-tile coordinates: 2 runs of 16 halves per tile ----
  const int arow   = tid >> 1;          // 0..127
  const int akbase = (tid & 1) * 32;    // 0 or 32
  const int am = m0 + arow;
  const bool avalid = (am < Mtot);
  int ihb = 0, iwb = 0, anH = 0;
  if (avalid) {
    int pq = a.P * a.Q;
    int an = am / pq;
    int rr = am - an * pq;
    int ap = rr / a.Q;
    int aq = rr - ap * a.Q;
    ihb = ap * a.stride - a.pad;
    iwb = aq * a.stride - a.pad;
    anH = an * a.H;
  }

  // ---- B-tile: two 16B asyncs per thread = 8 KB tile ----
  const int bkb  = tid >> 3;  // k row in tile (0..31; +32 for 2nd)
  const int bseg = tid & 7;   // 16B segment in 128B row
  const unsigned as_base = (unsigned)(size_t)(&As[0]);
  const unsigned bs_base = (unsigned)(size_t)(&Bs[0]);

  const int nkt = a.CRSp / KB;

  auto load_tile = [&](int kt, int buf) {
    const int k0 = kt * KB;
    if (FAST) {
#pragma unroll
      for (int ru = 0; ru < 2; ++ru) {
        // contiguous channel run: k = (r*S+s)*C + c, run inside one (r,s)
        int akoff = akbase + ru * 16;
        int krun = k0 + akoff;
        int c0 = krun & (a.C - 1);
        int rsi = krun >> a.clog2;
        int r = (int)(((unsigned)rsi * a.mulS) >> 16);
        int s = rsi - r * a.S;
        int ih = ihb + r;
        int iw = iwb + s;
        unsigned ldsA = as_base + (unsigned)((buf * BM * KB + arow * KB + akoff) * 2);
        if (avalid && (unsigned)ih < (unsigned)a.H && (unsigned)iw < (unsigned)a.W) {
          unsigned goff = (unsigned)((((long)(anH + ih) * a.W + iw) * a.C + c0) * 2);
          asm volatile("global_load_async_to_lds_b128 %0, %1, %2 offset:0"
                       :: "v"(ldsA), "v"(goff), "s"(a.X) : "memory");
          asm volatile("global_load_async_to_lds_b128 %0, %1, %2 offset:16"
                       :: "v"(ldsA), "v"(goff), "s"(a.X) : "memory");
        } else {
          v8h z = {};
          _Float16* dA = &As[buf * BM * KB + arow * KB + akoff];
          *(v8h*)dA = z;
          *(v8h*)(dA + 8) = z;
        }
      }
    } else {
      // stem path: per-element gather, magic-multiply decomposition (k < 2^14)
      _Float16* dA = &As[buf * BM * KB + arow * KB + akbase];
#pragma unroll
      for (int kk = 0; kk < 32; ++kk) {
        int k = k0 + akbase + kk;
        _Float16 v = (_Float16)0.0f;
        if (avalid && k < a.CRS) {
          int rsi = (int)(((unsigned)k * a.mulC) >> 16);
          int c = k - rsi * a.C;
          int r = (int)(((unsigned)rsi * a.mulS) >> 16);
          int s = rsi - r * a.S;
          int ih = ihb + r;
          int iw = iwb + s;
          if ((unsigned)ih < (unsigned)a.H && (unsigned)iw < (unsigned)a.W)
            v = a.X[((long)(anH + ih) * a.W + iw) * a.C + c];
        }
        dA[kk] = v;
      }
    }
    // B tile: packed weights, always in-bounds (CRSp padded), 2x async/thread
#pragma unroll
    for (int rb = 0; rb < 2; ++rb) {
      int kb = bkb + rb * 32;
      unsigned ldsB = bs_base + (unsigned)((buf * KB * BN + kb * BN + bseg * 8) * 2);
      unsigned goff = (unsigned)((((k0 + kb) * a.K) + n0 + bseg * 8) * 2);
      asm volatile("global_load_async_to_lds_b128 %0, %1, %2 offset:0"
                   :: "v"(ldsB), "v"(goff), "s"(a.Wp) : "memory");
    }
  };

  // wave tiling: 8 waves = 4 (M) x 2 (N), each wave owns a 32x32 sub-tile
  const int wv   = tid >> 5;
  const int lane = tid & 31;
  const int wm   = wv & 3;
  const int wn   = wv >> 2;
  const int g    = lane >> 4;
  const int lm   = lane & 15;

  v8f acc[2][2] = {};

  load_tile(0, 0);
  int buf = 0;
  for (int kt = 0; kt < nkt; ++kt) {
    asm volatile("s_wait_asynccnt 0" ::: "memory");
    __syncthreads();
    if (kt + 1 < nkt) load_tile(kt + 1, buf ^ 1);

    const _Float16* Ab = &As[buf * BM * KB];
    const _Float16* Bb = &Bs[buf * KB * BN];

#pragma unroll
    for (int kk = 0; kk < 2; ++kk) {  // two 32-wide K sub-tiles -> 8 WMMAs
      v16h afr[2], bfr[2];
#pragma unroll
      for (int i = 0; i < 2; ++i) {
        int mrow = wm * 32 + i * 16 + lm;
        v8h lo = *(const v8h*)(Ab + mrow * KB + kk * 32 + g * 8);
        v8h hi = *(const v8h*)(Ab + mrow * KB + kk * 32 + 16 + g * 8);
        afr[i] = __builtin_shufflevector(lo, hi, 0, 1, 2, 3, 4, 5, 6, 7,
                                         8, 9, 10, 11, 12, 13, 14, 15);
      }
#pragma unroll
      for (int j = 0; j < 2; ++j)
#pragma unroll
        for (int s = 0; s < 16; ++s)
          bfr[j][s] = Bb[(kk * 32 + (s < 8 ? g * 8 + s : 16 + g * 8 + (s - 8))) * BN + wn * 32 + j * 16 + lm];

#pragma unroll
      for (int i = 0; i < 2; ++i)
#pragma unroll
        for (int j = 0; j < 2; ++j)
          acc[i][j] = __builtin_amdgcn_wmma_f32_16x16x32_f16(
              false, afr[i], false, bfr[j], (short)0, acc[i][j], false, false);
    }
    buf ^= 1;
  }

  // Epilogue: NHWC -> output offset is simply m*K + ko (no divisions).
#pragma unroll
  for (int i = 0; i < 2; ++i) {
#pragma unroll
    for (int j = 0; j < 2; ++j) {
      int nl = wn * 32 + j * 16 + lm;
      float sc = s_scale[nl];
      float bs = s_bias[nl];
      int ko = n0 + nl;
#pragma unroll
      for (int v = 0; v < 8; ++v) {
        int m = m0 + wm * 32 + i * 16 + g * 8 + v;
        if (m < Mtot) {
          long oidx = (long)m * a.K + ko;
          float val = acc[i][j][v] * sc + bs;
          if (a.Res) val += (float)a.Res[oidx];
          if (a.relu) val = fmaxf(val, 0.0f);
          a.Y[oidx] = (_Float16)val;
        }
      }
    }
  }
}

// ---------------------------------------------------------------------------
// Maxpool 3x3 s2 p1 (NHWC f16)
// ---------------------------------------------------------------------------
__global__ void maxpool_kernel(const _Float16* __restrict__ X, _Float16* __restrict__ Y,
                               int N, int C, int H, int W, int P, int Q) {
  int idx = blockIdx.x * blockDim.x + threadIdx.x;
  if (idx >= N * P * Q * C) return;
  int c = idx % C;
  int t = idx / C;
  int q = t % Q; t /= Q;
  int p = t % P;
  int n = t / P;
  float m = -3.4e38f;
  for (int r = 0; r < 3; ++r) {
    int ih = p * 2 - 1 + r;
    if ((unsigned)ih >= (unsigned)H) continue;
    for (int s = 0; s < 3; ++s) {
      int iw = q * 2 - 1 + s;
      if ((unsigned)iw >= (unsigned)W) continue;
      m = fmaxf(m, (float)X[((long)(n * H + ih) * W + iw) * C + c]);
    }
  }
  Y[idx] = (_Float16)m;
}

// Global average pool over 7x7 (NHWC) -> comb[:, 0:512] f32
__global__ void avgpool_kernel(const _Float16* __restrict__ X, float* __restrict__ comb,
                               int Nb, int K) {
  int idx = blockIdx.x * blockDim.x + threadIdx.x;
  if (idx >= Nb * K) return;
  int k = idx % K;
  int n = idx / K;
  float s = 0.0f;
  for (int i = 0; i < 49; ++i) s += (float)X[((long)n * 49 + i) * K + k];
  comb[n * 520 + k] = s * (1.0f / 49.0f);
}

// angles = sigmoid(feats @ Wq^T + bq) * pi ; one wave per (batch, out)
__global__ void fcq_kernel(const float* __restrict__ comb, const float* __restrict__ w,
                           const float* __restrict__ b, float* __restrict__ angles) {
  int blk = blockIdx.x;
  int bi = blk >> 3;
  int o  = blk & 7;
  int lane = threadIdx.x;
  float s = 0.0f;
  for (int i = lane; i < 512; i += 32) s += comb[bi * 520 + i] * w[o * 512 + i];
  for (int off = 16; off > 0; off >>= 1) s += __shfl_down(s, off, 32);
  if (lane == 0) {
    float v = s + b[o];
    angles[bi * 8 + o] = (1.0f / (1.0f + __expf(-v))) * 3.14159265358979f;
  }
}

// ---------------------------------------------------------------------------
// 8-qubit statevector sim: one wave per batch element, state in LDS
// qubit q <-> bit position (7-q)
// ---------------------------------------------------------------------------
__global__ void quantum_kernel(const float* __restrict__ angles, const float* __restrict__ qw,
                               float* __restrict__ comb) {
  int b = blockIdx.x;
  int lane = threadIdx.x;
  __shared__ float re[256];
  __shared__ float im[256];
  for (int i = lane; i < 256; i += 32) { re[i] = 0.0f; im[i] = 0.0f; }
  __syncthreads();
  if (lane == 0) re[0] = 1.0f;
  __syncthreads();

  for (int q = 0; q < 8; ++q) {  // RX(angles)
    float h = angles[b * 8 + q] * 0.5f;
    float c = __cosf(h), s = __sinf(h);
    int bp = 7 - q, bit = 1 << bp;
    for (int p = lane; p < 128; p += 32) {
      int i0 = (((p >> bp) << (bp + 1)) | (p & (bit - 1)));
      int i1 = i0 | bit;
      float a0r = re[i0], a0i = im[i0], a1r = re[i1], a1i = im[i1];
      re[i0] = c * a0r + s * a1i;  im[i0] = c * a0i - s * a1r;
      re[i1] = c * a1r + s * a0i;  im[i1] = c * a1i - s * a0r;
    }
    __syncthreads();
  }
  for (int l = 0; l < 5; ++l) {
    for (int q = 0; q < 8; ++q) {  // RY(qw)
      float h = qw[l * 8 + q] * 0.5f;
      float c = __cosf(h), s = __sinf(h);
      int bp = 7 - q, bit = 1 << bp;
      for (int p = lane; p < 128; p += 32) {
        int i0 = (((p >> bp) << (bp + 1)) | (p & (bit - 1)));
        int i1 = i0 | bit;
        float a0r = re[i0], a0i = im[i0], a1r = re[i1], a1i = im[i1];
        re[i0] = c * a0r - s * a1r;  im[i0] = c * a0i - s * a1i;
        re[i1] = s * a0r + c * a1r;  im[i1] = s * a0i + c * a1i;
      }
      __syncthreads();
    }
    for (int q = 0; q < 8; ++q) {  // CNOT ring
      int cq = 7 - q, tq = 7 - ((q + 1) & 7);
      int lo = cq < tq ? cq : tq;
      int hi = cq < tq ? tq : cq;
      for (int p = lane; p < 64; p += 32) {
        int x = p;
        x = ((x >> lo) << (lo + 1)) | (x & ((1 << lo) - 1));
        x = ((x >> hi) << (hi + 1)) | (x & ((1 << hi) - 1));
        int i = x | (1 << cq);
        int j = i | (1 << tq);
        float tr = re[i], ti = im[i];
        re[i] = re[j]; im[i] = im[j];
        re[j] = tr;    im[j] = ti;
      }
      __syncthreads();
    }
  }
  float z[8] = {0, 0, 0, 0, 0, 0, 0, 0};
  for (int i = lane; i < 256; i += 32) {
    float p = re[i] * re[i] + im[i] * im[i];
#pragma unroll
    for (int q = 0; q < 8; ++q) z[q] += ((i >> (7 - q)) & 1) ? -p : p;
  }
  for (int off = 16; off > 0; off >>= 1)
#pragma unroll
    for (int q = 0; q < 8; ++q) z[q] += __shfl_down(z[q], off, 32);
  if (lane == 0)
#pragma unroll
    for (int q = 0; q < 8; ++q) comb[b * 520 + 512 + q] = z[q];
}

__global__ void cls1_kernel(const float* __restrict__ comb, const float* __restrict__ w,
                            const float* __restrict__ b, float* __restrict__ hdn) {
  int bi = blockIdx.x, o = blockIdx.y, lane = threadIdx.x;
  float s = 0.0f;
  for (int i = lane; i < 520; i += 32) s += comb[bi * 520 + i] * w[o * 520 + i];
  for (int off = 16; off > 0; off >>= 1) s += __shfl_down(s, off, 32);
  if (lane == 0) hdn[bi * 256 + o] = fmaxf(s + b[o], 0.0f);
}

__global__ void cls2_kernel(const float* __restrict__ hdn, const float* __restrict__ w,
                            const float* __restrict__ b, float* __restrict__ out) {
  int bi = blockIdx.x, o = blockIdx.y, lane = threadIdx.x;
  float s = 0.0f;
  for (int i = lane; i < 256; i += 32) s += hdn[bi * 256 + i] * w[o * 256 + i];
  for (int off = 16; off > 0; off >>= 1) s += __shfl_down(s, off, 32);
  if (lane == 0) out[bi * 4 + o] = s + b[o];
}

// ---------------------------------------------------------------------------
// Host orchestration
// ---------------------------------------------------------------------------
extern "C" void kernel_launch(void* const* d_in, const int* in_sizes, int n_in,
                              void* d_out, int out_size, void* d_ws, size_t ws_size,
                              hipStream_t stream) {
  (void)out_size; (void)ws_size;
  const int XN = 64 * 3 * 224 * 224;
  int xi = (in_sizes[0] == XN) ? 0 : (n_in - 1);
  int pbase = (xi == 0) ? 1 : 0;
  const float* x_f32 = (const float*)d_in[xi];
  auto P = [&](int i) -> const float* { return (const float*)d_in[pbase + i]; };

  // alphabetical pytree order: cls1.{b,w}, cls2.{b,w}, fcq.{b,w}, qweights,
  // resnet.bn1.{beta,gamma,mean,var}, resnet.conv1, layer{1..4}.{b0,b1}
  // block: bn1(4), bn2(4), conv1, conv2 [, down_bn(4), down_conv]
  const int IDX_CLS1_B = 0, IDX_CLS1_W = 1, IDX_CLS2_B = 2, IDX_CLS2_W = 3;
  const int IDX_FCQ_B = 4, IDX_FCQ_W = 5, IDX_QW = 6;
  const int IDX_RBN1 = 7, IDX_RCONV1 = 11;
  int bb[4][2];
  {
    int cur = 12;
    for (int L = 0; L < 4; ++L)
      for (int B2 = 0; B2 < 2; ++B2) {
        bb[L][B2] = cur;
        cur += (L > 0 && B2 == 0) ? 15 : 10;
      }
  }

  char* ws = (char*)d_ws;
  size_t cur_off = 0;
  auto alloc = [&](size_t bytes) -> char* {
    cur_off = (cur_off + 255) & ~(size_t)255;
    char* p = ws + cur_off;
    cur_off += bytes;
    return p;
  };
  _Float16* xf16 = (_Float16*)alloc((size_t)XN * 2);
  _Float16* bufA = (_Float16*)alloc((size_t)64 * 64 * 112 * 112 * 2);
  _Float16* bufB = (_Float16*)alloc((size_t)64 * 64 * 56 * 56 * 2);
  _Float16* bufC = (_Float16*)alloc((size_t)64 * 64 * 56 * 56 * 2);
  float* comb   = (float*)alloc((size_t)64 * 520 * 4);
  float* angles = (float*)alloc((size_t)64 * 8 * 4);
  float* hdn    = (float*)alloc((size_t)64 * 256 * 4);

  auto pack_conv = [&](int widx, int K, int C, int R) -> _Float16* {
    int CRS = C * R * R;
    int CRSp = (CRS + KB - 1) & ~(KB - 1);
    _Float16* wp = (_Float16*)alloc((size_t)CRSp * K * 2);
    int tot = CRSp * K;
    repack_weight_kernel<<<(tot + 255) / 256, 256, 0, stream>>>(P(widx), wp, K, C, R, R, CRSp);
    return wp;
  };

  auto run_conv = [&](const _Float16* X, const _Float16* Wp, int bnbase,
                      const _Float16* Res, _Float16* Y,
                      int C, int H, int K, int st, int pd, int R, int relu) {
    ConvArgs ca;
    ca.X = X; ca.Wp = Wp;
    ca.beta = P(bnbase); ca.gamma = P(bnbase + 1);
    ca.mean = P(bnbase + 2); ca.var = P(bnbase + 3);
    ca.Res = Res; ca.Y = Y;
    ca.Nb = 64; ca.C = C; ca.H = H; ca.W = H; ca.K = K;
    ca.P = (H + 2 * pd - R) / st + 1; ca.Q = ca.P;
    ca.R = R; ca.S = R; ca.stride = st; ca.pad = pd;
    ca.CRS = C * R * R; ca.CRSp = (ca.CRS + KB - 1) & ~(KB - 1); ca.relu = relu;
    ca.mulC = (unsigned)((65536 + C - 1) / C);
    ca.mulS = (unsigned)((65536 + R - 1) / R);
    int lg = 0; while ((1 << lg) < C) ++lg;
    ca.clog2 = lg;
    int Mtot = 64 * ca.P * ca.Q;
    dim3 grid((Mtot + BM - 1) / BM, K / BN);
    bool fast = (C >= 32) && ((C & (C - 1)) == 0);
    if (fast) conv_wmma_kernel<true><<<grid, 256, 0, stream>>>(ca);
    else      conv_wmma_kernel<false><<<grid, 256, 0, stream>>>(ca);
  };

  // 1) x -> NHWC f16
  cvt_nchw_to_nhwc_kernel<<<(XN + 255) / 256, 256, 0, stream>>>(x_f32, xf16);

  // 2) stem conv 7x7 s2 -> A(112), maxpool -> B(56)
  {
    _Float16* wp = pack_conv(IDX_RCONV1, 64, 3, 7);
    run_conv(xf16, wp, IDX_RBN1, nullptr, bufA, 3, 224, 64, 2, 3, 7, 1);
    int tot = 64 * 56 * 56 * 64;
    maxpool_kernel<<<(tot + 255) / 256, 256, 0, stream>>>(bufA, bufB, 64, 64, 112, 112, 56, 56);
  }

  // 3) 4 stages x 2 basic blocks (NHWC activations throughout)
  _Float16* in = bufB;
  int hw = 56, ci = 64;
  for (int L = 0; L < 4; ++L) {
    int co = 64 << L;
    int st = L ? 2 : 1;
    int oh = L ? hw / 2 : hw;
    int o0 = bb[L][0], o1 = bb[L][1];
    _Float16* other = (in == bufA) ? bufB : bufA;

    const _Float16* res0;
    _Float16* out0;
    if (L) {
      _Float16* wpd = pack_conv(o0 + 14, co, ci, 1);
      run_conv(in, wpd, o0 + 10, nullptr, other, ci, hw, co, 2, 0, 1, 0);
      res0 = other; out0 = in;
    } else {
      res0 = in; out0 = other;
    }
    _Float16* wp1 = pack_conv(o0 + 8, co, ci, 3);
    run_conv(in, wp1, o0 + 0, nullptr, bufC, ci, hw, co, st, 1, 3, 1);
    _Float16* wp2 = pack_conv(o0 + 9, co, co, 3);
    run_conv(bufC, wp2, o0 + 4, res0, out0, co, oh, co, 1, 1, 3, 1);

    _Float16* out1 = (out0 == bufA) ? bufB : bufA;
    _Float16* wp3 = pack_conv(o1 + 8, co, co, 3);
    run_conv(out0, wp3, o1 + 0, nullptr, bufC, co, oh, co, 1, 1, 3, 1);
    _Float16* wp4 = pack_conv(o1 + 9, co, co, 3);
    run_conv(bufC, wp4, o1 + 4, out0, out1, co, oh, co, 1, 1, 3, 1);

    in = out1; hw = oh; ci = co;
  }

  // 4) avgpool -> comb[:, :512]
  {
    int tot = 64 * 512;
    avgpool_kernel<<<(tot + 255) / 256, 256, 0, stream>>>(in, comb, 64, 512);
  }
  // 5) fcq -> angles ; quantum -> comb[:, 512:]
  fcq_kernel<<<64 * 8, 32, 0, stream>>>(comb, P(IDX_FCQ_W), P(IDX_FCQ_B), angles);
  quantum_kernel<<<64, 32, 0, stream>>>(angles, P(IDX_QW), comb);
  // 6) classifier
  cls1_kernel<<<dim3(64, 256), 32, 0, stream>>>(comb, P(IDX_CLS1_W), P(IDX_CLS1_B), hdn);
  cls2_kernel<<<dim3(64, 4), 32, 0, stream>>>(hdn, P(IDX_CLS2_W), P(IDX_CLS2_B), (float*)d_out);
}
